// MoE_84361747628175
// MI455X (gfx1250) — compile-verified, matrix operand order checked by reading, not run on hardware
//
#include <hip/hip_runtime.h>
#include <hip/hip_bf16.h>

// ---------------------------------------------------------------------------
// MoE forward for MI455X (gfx1250):
//   1) zero output + expert counters
//   2) (main path) convert w1/w2 fp32 -> bf16 into workspace (one pass; the
//      full bf16 weight set = 128 MB fits in the 192 MB L2)
//   3) router: softmax top-2, compact tokens into per-expert lists
//   4) per-expert fused MLP with v_wmma_f32_16x16x32_bf16; all fragments are
//      loaded as two contiguous b128 chunks; lds_h double-buffered so each
//      H-chunk costs a single workgroup barrier.
// ---------------------------------------------------------------------------

typedef __attribute__((ext_vector_type(16))) __bf16       v16bf;
typedef __attribute__((ext_vector_type(8)))  float        v8f;
typedef __attribute__((ext_vector_type(4)))  unsigned int u32x4;
typedef __attribute__((ext_vector_type(8)))  unsigned int u32x8;

constexpr int DM     = 1024;      // DMODEL
constexpr int NE     = 8;         // N_EXPERTS
constexpr int HS     = 4096;      // EXPERT_SIZE
constexpr int TOKENS = 2 * 2048;  // B*S
constexpr int TILE_M = 16;        // tokens per workgroup tile
constexpr int HCHUNK = 128;       // hidden-dim chunk (8 waves x 16 cols)
constexpr int XPITCH = DM + 8;    // LDS row pitch (bf16); 2064B: 16B-aligned,
                                  // 516 dwords == 4 mod 64 banks (2-way floor)
constexpr int HPITCH = HCHUNK + 8;// 272B = 16B-aligned

__device__ __forceinline__ __bf16 f2bf(float f) {
  unsigned u = __builtin_bit_cast(unsigned, f);
  unsigned short h = (unsigned short)(u >> 16);
  return __builtin_bit_cast(__bf16, h);
}

// Pack the bf16 truncations of two f32 into one dword with one v_perm_b32.
__device__ __forceinline__ unsigned pack_bf16(float lo, float hi) {
  return __builtin_amdgcn_perm(__builtin_bit_cast(unsigned, hi),
                               __builtin_bit_cast(unsigned, lo), 0x07060302u);
}

// --- WMMA 16-bit fragment loaders -----------------------------------------
// ISA 7.12.2: per lane the fragment is two contiguous 8-element (16B) runs:
//   elems 0..7  = row[k0 .. k0+7],   k0 = half*8
//   elems 8..15 = row[k0+16 .. +23]
// -> two b128 loads, zero shuffle VALU. Row (M for A, N for B) is selected by
// the caller via the row pointer; works for LDS (__shared__) or global bf16.
__device__ __forceinline__ v16bf frag_w(const __bf16* __restrict__ p, int half) {
  const u32x4* q0 = (const u32x4*)(p + half * 8);
  const u32x4* q1 = (const u32x4*)(p + 16 + half * 8);
  u32x4 a = *q0, b = *q1;
  u32x8 o;
  o[0] = a[0]; o[1] = a[1]; o[2] = a[2]; o[3] = a[3];
  o[4] = b[0]; o[5] = b[1]; o[6] = b[2]; o[7] = b[3];
  return __builtin_bit_cast(v16bf, o);
}

// Fallback: build the same fragment from f32 memory (4x b128 + 8x v_perm).
__device__ __forceinline__ v16bf frag_w(const float* __restrict__ p, int half) {
  const u32x4* q = (const u32x4*)(p + half * 8);
  const u32x4* r = (const u32x4*)(p + 16 + half * 8);
  u32x4 a0 = q[0], a1 = q[1], b0 = r[0], b1 = r[1];
  u32x8 o;
  o[0] = __builtin_amdgcn_perm(a0[1], a0[0], 0x07060302u);
  o[1] = __builtin_amdgcn_perm(a0[3], a0[2], 0x07060302u);
  o[2] = __builtin_amdgcn_perm(a1[1], a1[0], 0x07060302u);
  o[3] = __builtin_amdgcn_perm(a1[3], a1[2], 0x07060302u);
  o[4] = __builtin_amdgcn_perm(b0[1], b0[0], 0x07060302u);
  o[5] = __builtin_amdgcn_perm(b0[3], b0[2], 0x07060302u);
  o[6] = __builtin_amdgcn_perm(b1[1], b1[0], 0x07060302u);
  o[7] = __builtin_amdgcn_perm(b1[3], b1[2], 0x07060302u);
  return __builtin_bit_cast(v16bf, o);
}

// ---------------------------------------------------------------------------
// Kernel 0: zero output accumulator + per-expert counters
// ---------------------------------------------------------------------------
__global__ void zero_kernel(float* __restrict__ out, int n, int* __restrict__ counts) {
  int i = blockIdx.x * blockDim.x + threadIdx.x;
  if (i < n) out[i] = 0.0f;
  if (i < NE) counts[i] = 0;
}

// ---------------------------------------------------------------------------
// Kernel 0b: fp32 -> bf16 weight conversion, 8 elements / thread.
// ---------------------------------------------------------------------------
__global__ void convert_bf16_kernel(const u32x4* __restrict__ s, u32x4* __restrict__ d, int n8) {
  int i = blockIdx.x * blockDim.x + threadIdx.x;
  if (i < n8) {
    u32x4 a = s[2 * i], b = s[2 * i + 1];
    u32x4 o;
    o[0] = __builtin_amdgcn_perm(a[1], a[0], 0x07060302u);
    o[1] = __builtin_amdgcn_perm(a[3], a[2], 0x07060302u);
    o[2] = __builtin_amdgcn_perm(b[1], b[0], 0x07060302u);
    o[3] = __builtin_amdgcn_perm(b[3], b[2], 0x07060302u);
    d[i] = o;
  }
}

// ---------------------------------------------------------------------------
// Kernel 1: router. One wave32 per token: logits = x . gate_w^T, softmax,
// top-2, append (token, prob) to compacted per-expert lists.
// ---------------------------------------------------------------------------
__global__ __launch_bounds__(256) void router_kernel(
    const float* __restrict__ x, const float* __restrict__ gw,
    int* __restrict__ counts, int* __restrict__ etok, float* __restrict__ ewgt) {
  const int lane = threadIdx.x & 31;
  const int wv   = threadIdx.x >> 5;
  const int t    = blockIdx.x * 8 + wv;
  if (t >= TOKENS) return;

  float xv[32];
  const float* xr = x + (size_t)t * DM + lane * 32;
#pragma unroll
  for (int j = 0; j < 32; ++j) xv[j] = xr[j];

  float logit[NE];
#pragma unroll
  for (int e = 0; e < NE; ++e) {
    const float* gr = gw + e * DM + lane * 32;
    float s = 0.0f;
#pragma unroll
    for (int j = 0; j < 32; ++j) s = fmaf(xv[j], gr[j], s);
#pragma unroll
    for (int off = 16; off > 0; off >>= 1) s += __shfl_xor(s, off, 32);
    logit[e] = s;
  }

  if (lane == 0) {
    float mx = logit[0];
#pragma unroll
    for (int e = 1; e < NE; ++e) mx = fmaxf(mx, logit[e]);
    float p[NE], sum = 0.0f;
#pragma unroll
    for (int e = 0; e < NE; ++e) { p[e] = __expf(logit[e] - mx); sum += p[e]; }
    const float inv = 1.0f / sum;
    int i0 = 0; float p0 = p[0];
#pragma unroll
    for (int e = 1; e < NE; ++e) if (p[e] > p0) { p0 = p[e]; i0 = e; }
    int i1 = -1; float p1 = -1.0f;
#pragma unroll
    for (int e = 0; e < NE; ++e) if (e != i0 && p[e] > p1) { p1 = p[e]; i1 = e; }

    int pos0 = atomicAdd(&counts[i0], 1);
    etok[i0 * TOKENS + pos0] = t;
    ewgt[i0 * TOKENS + pos0] = p0 * inv;
    int pos1 = atomicAdd(&counts[i1], 1);
    etok[i1 * TOKENS + pos1] = t;
    ewgt[i1 * TOKENS + pos1] = p1 * inv;
  }
}

// ---------------------------------------------------------------------------
// Kernel 2: per-expert fused MLP on compacted tokens.
// Grid: (TOKENS/TILE_M tile-slots, NE experts), block = 256 (8 waves).
// WT = __bf16 (pre-converted weights in ws) or float (fallback, inline perm).
// ---------------------------------------------------------------------------
template <typename WT>
__global__ __launch_bounds__(256) void moe_expert_kernel(
    const float* __restrict__ x,
    const WT* __restrict__ w1, const float* __restrict__ b1,
    const WT* __restrict__ w2, const float* __restrict__ b2,
    const int* __restrict__ counts, const int* __restrict__ etok,
    const float* __restrict__ ewgt, float* __restrict__ out) {
  __shared__ alignas(16) __bf16 lds_x[TILE_M * XPITCH];      // gathered tokens
  __shared__ alignas(16) __bf16 lds_h[2][TILE_M * HPITCH];   // relu(h), dbl-buf
  __shared__ int    s_tok[TILE_M];
  __shared__ float  s_w[TILE_M];

  const int e    = blockIdx.y;
  const int cnt  = counts[e];
  const int base = blockIdx.x * TILE_M;
  if (base >= cnt) return;  // empty tile slot for this expert

  const int tid  = threadIdx.x;
  const int lane = tid & 31;
  const int wv   = tid >> 5;    // wave 0..7: owns output cols [wv*128, +128)
  const int half = lane >> 4;
  const int ln16 = lane & 15;

  if (tid < TILE_M) {
    int slot = base + tid;
    if (slot < cnt) {
      s_tok[tid] = etok[e * TOKENS + slot];
      s_w[tid]   = ewgt[e * TOKENS + slot];
    } else {
      s_tok[tid] = -1;
      s_w[tid]   = 0.0f;
    }
  }
  __syncthreads();

  // Gather x rows, fp32 -> packed bf16 pairs (1 v_perm + 1 dword store / pair)
  {
    unsigned* ldsx32 = (unsigned*)lds_x;
    for (int i = tid; i < TILE_M * (DM / 2); i += 256) {
      int row = i >> 9, c2 = i & (DM / 2 - 1);
      int t   = s_tok[row];
      unsigned v = 0u;
      if (t >= 0) {
        const float* px = x + (size_t)t * DM + c2 * 2;
        v = pack_bf16(px[0], px[1]);
      }
      ldsx32[row * (XPITCH / 2) + c2] = v;
    }
  }
  __syncthreads();

  const WT*    w1e = w1 + (size_t)e * HS * DM;
  const float* b1e = b1 + (size_t)e * HS;
  const WT*    w2e = w2 + (size_t)e * DM * HS;
  const float* b2e = b2 + (size_t)e * DM;

  v8f acc[8];  // 8 x (16x16 f32) output tiles per wave = 128 columns
#pragma unroll
  for (int nt = 0; nt < 8; ++nt)
#pragma unroll
    for (int r = 0; r < 8; ++r) acc[nt][r] = 0.0f;

  const __bf16* xrow = lds_x + ln16 * XPITCH;  // A row: token M = lane&15

  // Per-nt w2 row bases (B row: N = output column d)
  const WT* w2row[8];
#pragma unroll
  for (int nt = 0; nt < 8; ++nt)
    w2row[nt] = w2e + (size_t)(wv * 128 + nt * 16 + ln16) * HS;

  int pp = 0;  // lds_h parity
  for (int hh = 0; hh < HS; hh += HCHUNK, pp ^= 1) {
    // ---- Phase 1: h[:, hh+wv*16 .. +16] = relu(x.w1^T + b1); K = DM.
    const int hbase = hh + wv * 16;
    v8f acch;
#pragma unroll
    for (int r = 0; r < 8; ++r) acch[r] = 0.0f;

    const WT* w1row = w1e + (size_t)(hbase + ln16) * DM;
#pragma unroll 4
    for (int kk = 0; kk < DM; kk += 32) {
      v16bf a = frag_w(xrow + kk, half);
      v16bf b = frag_w(w1row + kk, half);
      acch = __builtin_amdgcn_wmma_f32_16x16x32_bf16(false, a, false, b,
                                                     (short)0, acch, false, false);
    }
    const float bb1 = b1e[hbase + ln16];
#pragma unroll
    for (int r = 0; r < 8; ++r) {
      int m = r + half * 8;                    // C/D layout: M = vgpr + 8*half
      lds_h[pp][m * HPITCH + wv * 16 + ln16] = f2bf(fmaxf(acch[r] + bb1, 0.0f));
    }
    // Single barrier per chunk: __syncthreads waits dscnt==0 before signaling,
    // so chunk-c reads finish before any wave can start writing buffer pp
    // again at chunk c+2 (gated behind barrier c+1). Buffer pp^1 is untouched
    // by phase 2 below, so no trailing barrier is needed.
    __syncthreads();

    // ---- Phase 2: acc += relu(h)_chunk . w2^T for this wave's 128 columns.
#pragma unroll
    for (int hk = 0; hk < HCHUNK / 32; ++hk) {
      v16bf a = frag_w(&lds_h[pp][ln16 * HPITCH + hk * 32], half);
#pragma unroll
      for (int nt = 0; nt < 8; ++nt) {
        v16bf b = frag_w(w2row[nt] + hh + hk * 32, half);
        acc[nt] = __builtin_amdgcn_wmma_f32_16x16x32_bf16(false, a, false, b,
                                                          (short)0, acc[nt], false, false);
      }
    }
  }

  // ---- Epilogue: out[t,d] += route_w * (acc + b2[d]); hw f32 atomic add.
  //      Each (t,d) gets exactly 2 commutative adds -> deterministic.
#pragma unroll
  for (int nt = 0; nt < 8; ++nt) {
    int   d  = wv * 128 + nt * 16 + ln16;
    float bb = b2e[d];
#pragma unroll
    for (int r = 0; r < 8; ++r) {
      int m = r + half * 8;
      int t = s_tok[m];
      if (t >= 0) {
        unsafeAtomicAdd(&out[(size_t)t * DM + d], (acc[nt][r] + bb) * s_w[m]);
      }
    }
  }
}

// ---------------------------------------------------------------------------
// Host launcher (graph-capture safe: only stream launches)
// ---------------------------------------------------------------------------
extern "C" void kernel_launch(void* const* d_in, const int* in_sizes, int n_in,
                              void* d_out, int out_size, void* d_ws, size_t ws_size,
                              hipStream_t stream) {
  const float* x      = (const float*)d_in[0];
  const float* gate_w = (const float*)d_in[1];
  const float* w1     = (const float*)d_in[2];
  const float* b1     = (const float*)d_in[3];
  const float* w2     = (const float*)d_in[4];
  const float* b2     = (const float*)d_in[5];
  float*       out    = (float*)d_out;

  const size_t wElems  = (size_t)NE * HS * DM;        // per weight tensor
  const size_t wBytes  = wElems * sizeof(__bf16);     // 64 MB
  const size_t listOff = 2 * wBytes;                  // after w1b, w2b
  const size_t needBig = listOff + 256 + (size_t)NE * TOKENS * (sizeof(int) + sizeof(float));

  const int  n_out = TOKENS * DM;
  const dim3 gemm_grid(TOKENS / TILE_M, NE);

  char* ws = (char*)d_ws;

  if (ws_size >= needBig) {
    // Main path: bf16 weights staged in workspace (fits in 192 MB L2).
    __bf16* w1b    = (__bf16*)ws;
    __bf16* w2b    = (__bf16*)(ws + wBytes);
    int*    counts = (int*)(ws + listOff);
    int*    etok   = (int*)(ws + listOff + 256);
    float*  ewgt   = (float*)(ws + listOff + 256 + (size_t)NE * TOKENS * sizeof(int));

    const int n8 = (int)(wElems / 8);
    convert_bf16_kernel<<<(n8 + 255) / 256, 256, 0, stream>>>(
        (const u32x4*)w1, (u32x4*)w1b, n8);
    convert_bf16_kernel<<<(n8 + 255) / 256, 256, 0, stream>>>(
        (const u32x4*)w2, (u32x4*)w2b, n8);

    zero_kernel<<<(n_out + 255) / 256, 256, 0, stream>>>(out, n_out, counts);
    router_kernel<<<TOKENS / 8, 256, 0, stream>>>(x, gate_w, counts, etok, ewgt);
    moe_expert_kernel<__bf16><<<gemm_grid, 256, 0, stream>>>(
        x, w1b, b1, w2b, b2, counts, etok, ewgt, out);
  } else {
    // Fallback: stream fp32 weights, convert in-register via v_perm.
    int*   counts = (int*)ws;
    int*   etok   = (int*)(ws + 256);
    float* ewgt   = (float*)(ws + 256 + (size_t)NE * TOKENS * sizeof(int));

    zero_kernel<<<(n_out + 255) / 256, 256, 0, stream>>>(out, n_out, counts);
    router_kernel<<<TOKENS / 8, 256, 0, stream>>>(x, gate_w, counts, etok, ewgt);
    moe_expert_kernel<float><<<gemm_grid, 256, 0, stream>>>(
        x, w1, b1, w2, b2, counts, etok, ewgt, out);
  }
}
